// LSTMOtm_21663815041518
// MI455X (gfx1250) — matrix operand order repaired and checked by reading, not compiled, not run on hardware
//
#include <hip/hip_runtime.h>
#include <hip/hip_bf16.h>

// Problem constants (from reference): B,T,E,D,V
#define BB   64
#define TT   32
#define EE   512
#define DD   512
#define VV   32000
#define GG   2048        // 4*D
#define TM1  31          // T-1 decode steps

typedef __attribute__((ext_vector_type(8)))  __bf16 v8bf;
typedef __attribute__((ext_vector_type(16))) __bf16 v16bf;
typedef __attribute__((ext_vector_type(8)))  float  v8f;

union BF16x16 { v16bf v; v8bf h[2]; };

__device__ __forceinline__ v8f wmma_bf16(v16bf a, v16bf b, v8f c) {
    // D = A(16x32 bf16) x B(32x16 bf16) + C(16x16 f32)
    return __builtin_amdgcn_wmma_f32_16x16x32_bf16(
        /*neg_a=*/false, a, /*neg_b=*/false, b,
        /*c_mod=*/(short)0, c, /*reuse_a=*/false, /*reuse_b=*/false);
}

// ---------------------------------------------------------------------------
// Kernel 1: stable argsort of caption lengths, descending (= argsort(-lens)).
// ---------------------------------------------------------------------------
__global__ void sort_kernel(const int* __restrict__ caplen,
                            int* __restrict__ sidx, int* __restrict__ dlen,
                            int* __restrict__ out_tail) {
    int p = threadIdx.x;
    if (p >= BB) return;
    int lp = caplen[p];
    int rank = 0;
    for (int j = 0; j < BB; ++j) {
        int lj = caplen[j];
        rank += (lj > lp) || (lj == lp && j < p);
    }
    sidx[rank] = p;
    dlen[rank] = lp - 1;
    out_tail[rank] = p;   // sort_ind written as int bits into d_out tail
}

// ---------------------------------------------------------------------------
// Kernel 2: one-pass setup: bf16 conversions, encoder gather by sort order,
// bias sum, zero h/c state.
// ---------------------------------------------------------------------------
__global__ void setup_convert(const float* __restrict__ fc_w,
                              const float* __restrict__ W_ih,
                              const float* __restrict__ W_hh,
                              const float* __restrict__ b_ih,
                              const float* __restrict__ b_hh,
                              const float* __restrict__ enc,
                              const int*   __restrict__ sidx,
                              __bf16* __restrict__ fcb,
                              __bf16* __restrict__ wih,
                              __bf16* __restrict__ whh,
                              float*  __restrict__ bias,
                              __bf16* __restrict__ encb,
                              float*  __restrict__ cst,
                              __bf16* __restrict__ hb) {
    size_t i = (size_t)blockIdx.x * blockDim.x + threadIdx.x;
    const size_t NFC = (size_t)VV * DD;
    const size_t NW  = (size_t)GG * EE;
    const size_t NE  = (size_t)BB * TT * EE;
    const size_t NS  = (size_t)BB * DD;
    if (i < NFC) { fcb[i] = (__bf16)fc_w[i]; return; }
    i -= NFC;
    if (i < NW)  { wih[i] = (__bf16)W_ih[i]; return; }
    i -= NW;
    if (i < NW)  { whh[i] = (__bf16)W_hh[i]; return; }
    i -= NW;
    if (i < NE)  {
        size_t bpos = i / (TT * EE);
        size_t rem  = i - bpos * (TT * EE);
        encb[i] = (__bf16)enc[(size_t)sidx[bpos] * (TT * EE) + rem];
        return;
    }
    i -= NE;
    if (i < GG)  { bias[i] = b_ih[i] + b_hh[i]; return; }
    i -= GG;
    if (i < NS)  { cst[i] = 0.0f; return; }
    i -= NS;
    if (i < NS)  { hb[i] = (__bf16)0.0f; return; }
}

// ---------------------------------------------------------------------------
// Kernel 3 (per step): gates = x_t @ W_ih^T + h @ W_hh^T + (b_ih+b_hh)
// 16 blocks x 8 waves = 128 waves; one 16-col gate tile per wave.
// h staged in LDS; W fragments double-buffered from global.
// ---------------------------------------------------------------------------
__global__ __launch_bounds__(256)
void gates_kernel(const __bf16* __restrict__ encb,
                  const __bf16* __restrict__ hb,
                  const __bf16* __restrict__ wih,
                  const __bf16* __restrict__ whh,
                  const float*  __restrict__ bias,
                  float* __restrict__ gates, int t) {
    __shared__ __bf16 sh[BB * DD];                 // 64 KB: h state
    {
        const uint4* src = (const uint4*)hb;
        uint4*       dst = (uint4*)sh;
        int tid = threadIdx.x;
        #pragma unroll
        for (int i = 0; i < 16; ++i) dst[tid + i * 256] = src[tid + i * 256];
    }
    __syncthreads();

    const int tid   = threadIdx.x;
    const int wave  = tid >> 5;
    const int lane  = tid & 31;
    const int laneM = lane & 15;
    const int hi    = lane >> 4;
    const int n0    = (blockIdx.x * 8 + wave) * 16;    // gate column tile

    // Per-lane base pointers (all inner-loop offsets become immediates).
    const __bf16* bpx = wih + (size_t)(n0 + laneM) * EE + hi * 16;
    const __bf16* bph = whh + (size_t)(n0 + laneM) * DD + hi * 16;
    const __bf16* apx = encb + ((size_t)laneM * TT + t) * EE + hi * 16;  // + mi*16*TT*EE + k*32
    const __bf16* aph = sh + laneM * DD + hi * 16;                       // + mi*16*DD + k*32

    v8f acc[4] = {};
    BF16x16 bb[2];

    // ---- phase 1: x_t @ W_ih^T (A from global encb) ----
    bb[0].h[0] = *(const v8bf*)(bpx);
    bb[0].h[1] = *(const v8bf*)(bpx + 8);
    #pragma unroll
    for (int k = 0; k < 16; ++k) {
        const int cur = k & 1, nxt = cur ^ 1;
        if (k < 15) {
            const __bf16* q = bpx + (k + 1) * 32;
            bb[nxt].h[0] = *(const v8bf*)(q);
            bb[nxt].h[1] = *(const v8bf*)(q + 8);
        }
        #pragma unroll
        for (int mi = 0; mi < 4; ++mi) {
            BF16x16 a;
            const __bf16* p = apx + (size_t)mi * 16 * TT * EE + k * 32;
            a.h[0] = *(const v8bf*)(p);
            a.h[1] = *(const v8bf*)(p + 8);
            acc[mi] = wmma_bf16(a.v, bb[cur].v, acc[mi]);
        }
    }
    // ---- phase 2: h @ W_hh^T (A from LDS) ----
    bb[0].h[0] = *(const v8bf*)(bph);
    bb[0].h[1] = *(const v8bf*)(bph + 8);
    #pragma unroll
    for (int k = 0; k < 16; ++k) {
        const int cur = k & 1, nxt = cur ^ 1;
        if (k < 15) {
            const __bf16* q = bph + (k + 1) * 32;
            bb[nxt].h[0] = *(const v8bf*)(q);
            bb[nxt].h[1] = *(const v8bf*)(q + 8);
        }
        #pragma unroll
        for (int mi = 0; mi < 4; ++mi) {
            BF16x16 a;
            const __bf16* p = aph + mi * 16 * DD + k * 32;
            a.h[0] = *(const v8bf*)(p);
            a.h[1] = *(const v8bf*)(p + 8);
            acc[mi] = wmma_bf16(a.v, bb[cur].v, acc[mi]);
        }
    }

    const int n = n0 + laneM;
    const float bsum = bias[n];
    #pragma unroll
    for (int mi = 0; mi < 4; ++mi) {
        #pragma unroll
        for (int r = 0; r < 8; ++r) {
            int m = mi * 16 + r + hi * 8;           // sorted batch row
            gates[(size_t)m * GG + n] = acc[mi][r] + bsum;
        }
    }
}

// ---------------------------------------------------------------------------
// Kernel 4 (per step): elementwise LSTM cell with masked state update.
// ---------------------------------------------------------------------------
__global__ void lstm_cell(const float* __restrict__ gates,
                          float* __restrict__ c,
                          __bf16* __restrict__ hb,
                          const int* __restrict__ dlen, int t) {
    int i = blockIdx.x * blockDim.x + threadIdx.x;   // < 64*512
    int b = i >> 9;
    int d = i & 511;
    const float* g = gates + (size_t)b * GG;
    float gi = g[d], gf = g[DD + d], gg = g[2 * DD + d], go = g[3 * DD + d];
    float I = 1.0f / (1.0f + __expf(-gi));
    float F = 1.0f / (1.0f + __expf(-gf));
    float Gv = tanhf(gg);
    float O = 1.0f / (1.0f + __expf(-go));
    float cn = F * c[i] + I * Gv;
    float hn = O * tanhf(cn);
    if (t < dlen[b]) {              // active lanes update state
        c[i]  = cn;
        hb[i] = (__bf16)hn;
    }
}

// ---------------------------------------------------------------------------
// Kernel 5 (per step): preds = h @ fc_w^T + fc_b (zeroed for inactive rows).
// 125 blocks x 8 waves, TWO 16-col vocab tiles per wave (N=32), full M=64 via
// 4 accumulator pairs. h staged in LDS; fc_w fragments double-buffered.
// 8 independent v_wmma per k-iteration per wave.
// ---------------------------------------------------------------------------
__global__ __launch_bounds__(256)
void vocab_kernel(const __bf16* __restrict__ hb,
                  const __bf16* __restrict__ fcb,
                  const float*  __restrict__ fc_b,
                  const int*    __restrict__ dlen,
                  float* __restrict__ out, int t) {
    __shared__ __bf16 sh[BB * DD];                 // 64 KB: h state
    {
        const uint4* src = (const uint4*)hb;
        uint4*       dst = (uint4*)sh;
        int tid = threadIdx.x;
        #pragma unroll
        for (int i = 0; i < 16; ++i) dst[tid + i * 256] = src[tid + i * 256];
    }
    __syncthreads();

    const int tid   = threadIdx.x;
    const int wave  = tid >> 5;
    const int lane  = tid & 31;
    const int laneM = lane & 15;
    const int hi    = lane >> 4;
    const int n0    = (blockIdx.x * 8 + wave) * 32;     // two vocab column tiles

    const __bf16* bp0 = fcb + (size_t)(n0 + laneM) * DD + hi * 16;
    const __bf16* bp1 = bp0 + (size_t)16 * DD;          // tile n0+16
    const __bf16* ap  = sh + laneM * DD + hi * 16;       // + mi*16*DD + k*32

    v8f acc[4][2] = {};
    BF16x16 b0[2], b1[2];
    b0[0].h[0] = *(const v8bf*)(bp0);  b0[0].h[1] = *(const v8bf*)(bp0 + 8);
    b1[0].h[0] = *(const v8bf*)(bp1);  b1[0].h[1] = *(const v8bf*)(bp1 + 8);

    #pragma unroll
    for (int k = 0; k < 16; ++k) {
        const int cur = k & 1, nxt = cur ^ 1;
        if (k < 15) {
            const __bf16* q0 = bp0 + (k + 1) * 32;
            const __bf16* q1 = bp1 + (k + 1) * 32;
            b0[nxt].h[0] = *(const v8bf*)(q0);  b0[nxt].h[1] = *(const v8bf*)(q0 + 8);
            b1[nxt].h[0] = *(const v8bf*)(q1);  b1[nxt].h[1] = *(const v8bf*)(q1 + 8);
        }
        #pragma unroll
        for (int mi = 0; mi < 4; ++mi) {
            BF16x16 a;
            const __bf16* p = ap + mi * 16 * DD + k * 32;
            a.h[0] = *(const v8bf*)(p);
            a.h[1] = *(const v8bf*)(p + 8);
            acc[mi][0] = wmma_bf16(a.v, b0[cur].v, acc[mi][0]);
            acc[mi][1] = wmma_bf16(a.v, b1[cur].v, acc[mi][1]);
        }
    }

    #pragma unroll
    for (int nt = 0; nt < 2; ++nt) {
        const int n = n0 + nt * 16 + laneM;
        const float fb = fc_b[n];
        #pragma unroll
        for (int mi = 0; mi < 4; ++mi) {
            #pragma unroll
            for (int r = 0; r < 8; ++r) {
                int m = mi * 16 + r + hi * 8;           // sorted batch row
                float v = (t < dlen[m]) ? (acc[mi][nt][r] + fb) : 0.0f;
                out[((size_t)m * TM1 + t) * VV + n] = v;
            }
        }
    }
}

// ---------------------------------------------------------------------------
extern "C" void kernel_launch(void* const* d_in, const int* in_sizes, int n_in,
                              void* d_out, int out_size, void* d_ws, size_t ws_size,
                              hipStream_t stream) {
    const float* enc    = (const float*)d_in[0];   // (B,T,E)
    const int*   caplen = (const int*)  d_in[1];   // (B,1)
    const float* W_ih   = (const float*)d_in[2];   // (4D,E)
    const float* W_hh   = (const float*)d_in[3];   // (4D,D)
    const float* b_ih   = (const float*)d_in[4];   // (4D,)
    const float* b_hh   = (const float*)d_in[5];   // (4D,)
    const float* fc_w   = (const float*)d_in[6];   // (V,D)
    const float* fc_b   = (const float*)d_in[7];   // (V,)
    float* out = (float*)d_out;

    // Workspace carve-out (256B aligned regions), ~40 MB total.
    char* ws = (char*)d_ws;
    size_t off = 0;
    #define WS_ALLOC(ptrty, name, bytes) \
        ptrty name = (ptrty)(ws + off); off = (off + (size_t)(bytes) + 255) & ~(size_t)255;
    WS_ALLOC(__bf16*, fcb,  (size_t)VV * DD * 2)        // fc_w bf16
    WS_ALLOC(__bf16*, wih,  (size_t)GG * EE * 2)        // W_ih bf16
    WS_ALLOC(__bf16*, whh,  (size_t)GG * DD * 2)        // W_hh bf16
    WS_ALLOC(__bf16*, encb, (size_t)BB * TT * EE * 2)   // sorted encoder bf16
    WS_ALLOC(float*,  bias, (size_t)GG * 4)             // b_ih + b_hh
    WS_ALLOC(float*,  gates,(size_t)BB * GG * 4)        // per-step gates
    WS_ALLOC(__bf16*, hb,   (size_t)BB * DD * 2)        // h state (bf16)
    WS_ALLOC(float*,  cst,  (size_t)BB * DD * 4)        // c state (fp32)
    WS_ALLOC(int*,    dlen, (size_t)BB * 4)
    WS_ALLOC(int*,    sidx, (size_t)BB * 4)
    #undef WS_ALLOC
    (void)in_sizes; (void)n_in; (void)out_size; (void)ws_size;

    // sort_ind goes into the output tail, as int bits after predictions.
    int* out_tail = (int*)out + (size_t)BB * TM1 * VV;

    sort_kernel<<<1, 64, 0, stream>>>(caplen, sidx, dlen, out_tail);

    const size_t total = (size_t)VV * DD + 2 * (size_t)GG * EE
                       + (size_t)BB * TT * EE + GG + 2 * (size_t)BB * DD;
    const int cvt_blocks = (int)((total + 255) / 256);
    setup_convert<<<cvt_blocks, 256, 0, stream>>>(fc_w, W_ih, W_hh, b_ih, b_hh,
                                                  enc, sidx, fcb, wih, whh,
                                                  bias, encb, cst, hb);

    for (int t = 0; t < TM1; ++t) {
        gates_kernel<<<16, 256, 0, stream>>>(encb, hb, wih, whh, bias, gates, t);
        lstm_cell<<<(BB * DD) / 256, 256, 0, stream>>>(gates, cst, hb, dlen, t);
        vocab_kernel<<<VV / (32 * 8), 256, 0, stream>>>(hb, fcb, fc_b, dlen, out, t);
    }
}
